// ConvMem_45440753992280
// MI455X (gfx1250) — matrix operand order repaired and checked by reading, not compiled, not run on hardware
//
#include <hip/hip_runtime.h>

typedef __attribute__((ext_vector_type(16))) __bf16          v16bf;
typedef __attribute__((ext_vector_type(16))) unsigned short  v16us;
typedef __attribute__((ext_vector_type(8)))  unsigned short  v8us;
typedef __attribute__((ext_vector_type(8)))  float           v8f;
typedef __attribute__((ext_vector_type(4)))  unsigned int    u32x4;
typedef __attribute__((ext_vector_type(8)))  unsigned int    u32x8;

#define BDIM 64
#define CX   16
#define CY   64
#define LL   32
#define KKK  5
#define NN   1024
#define MMM  28
#define POSN (MMM*MMM)        // 784
#define KF   (CY*POSN)        // 50176
#define ETAX 0.1f             // DT/TAU_X
#define ETAY 1.0f             // DT/TAU_Y
#define STEPS 50
#define ZSPLIT 16             // split-K factor for GEMM1
#define YTN  128              // ko columns per gemm_y block

__device__ __forceinline__ unsigned short f2bf(float x) {
  unsigned int u = __float_as_uint(x);
  u += 0x7fffu + ((u >> 16) & 1u);     // round-to-nearest-even
  return (unsigned short)(u >> 16);
}

// 32-bit LDS byte offset of a shared-memory object (generic -> AS3 addrspacecast)
__device__ __forceinline__ unsigned lds_off_u32(const void* p) {
  return (unsigned)(size_t)(const __attribute__((address_space(3))) char*)p;
}

// ---- Tensor Data Mover: async DMA of a 32-row x 128-col bf16 tile of W into LDS ----
// D# per CDNA5 ISA: group0 {count,lds,global,type=2},
// group1 {data_size=2B, tensor_dim0=KF, tile_dim0=128, tile_dim1=32, dim0_stride=KF}.
__device__ __forceinline__ void tdm_load_w_tile(unsigned lds_byte, const unsigned short* gptr) {
  unsigned long long ga = (unsigned long long)(size_t)gptr;
  u32x4 g0 = { 1u,                                        // count=1 (valid user D#)
               lds_byte,                                  // lds_addr
               (unsigned)ga,                              // global_addr[31:0]
               (unsigned)((ga >> 32) & 0x01ffffffu) | (2u << 30) };  // addr[56:32] | type=2
  u32x8 g1 = { 0x00010000u,                               // data_size=1 (2 bytes), no flags
               (unsigned)((KF & 0xffffu) << 16),          // tensor_dim0[15:0] << 16
               0u,                                        // tensor_dim0[31:16] | tensor_dim1.lo<<16
               16u | (128u << 16),                        // tensor_dim1.hi | tile_dim0=128
               32u,                                       // tile_dim1=32, tile_dim2=0
               (unsigned)KF,                              // tensor_dim0_stride[31:0]
               0u,                                        // stride hi | dim1_stride.lo
               0u };
  asm volatile("tensor_load_to_lds %0, %1" :: "s"(g0), "s"(g1) : "memory");
}

// ---------------- elementwise helpers ----------------
__global__ __launch_bounds__(256) void cvt_bf_kernel(const float* __restrict__ src,
                                                     unsigned short* __restrict__ dst, int n) {
  int i = blockIdx.x * 256 + threadIdx.x;
  if (i < n) dst[i] = f2bf(src[i]);
}

__global__ __launch_bounds__(256) void copy_f32_kernel(const float* __restrict__ src,
                                                       float* __restrict__ dst, int n) {
  int i = blockIdx.x * 256 + threadIdx.x;
  if (i < n) dst[i] = src[i];
}

// -------- GEMM1 (split-K): zp[p][b,n] = sum_{k in chunk p} f[b,k] * W[n,k] --------
__global__ __launch_bounds__(32) void gemm_z_kernel(const unsigned short* __restrict__ Wb,
                                                    const unsigned short* __restrict__ Fb,
                                                    float* __restrict__ zp) {
  const int lane = threadIdx.x;
  const int m15 = lane & 15, hi = lane >> 4;
  const int kbase = blockIdx.z * (KF / ZSPLIT);           // 3136 per chunk
  const unsigned short* arow = Fb + (size_t)(blockIdx.x * 16 + m15) * KF + kbase;
  const unsigned short* brow = Wb + (size_t)(blockIdx.y * 16 + m15) * KF + kbase;
  v8f acc = {};
  for (int k0 = 0; k0 < KF / ZSPLIT; k0 += 32) {
    v16us a, bm;
    a = *(const v16us*)(arow + k0 + hi * 16);                     // K hi*16 .. +15
    bm = *(const v16us*)(brow + k0 + hi * 16);                    // K hi*16 .. +15
    __builtin_prefetch(brow + k0 + 512, 0, 1);                    // global_prefetch_b8
    acc = __builtin_amdgcn_wmma_f32_16x16x32_bf16(
        false, __builtin_bit_cast(v16bf, a),
        false, __builtin_bit_cast(v16bf, bm),
        (short)0, acc, false, false);
  }
  float* zo = zp + (size_t)blockIdx.z * (BDIM * NN);
  const int zb = blockIdx.x * 16, zn = blockIdx.y * 16;
#pragma unroll
  for (int r = 0; r < 8; ++r)
    zo[(size_t)(zb + r + 8 * hi) * NN + zn + m15] = acc[r];
}

// deterministic split-K reduction
__global__ __launch_bounds__(256) void reduce_z_kernel(const float* __restrict__ zp,
                                                       float* __restrict__ z) {
  const int i = blockIdx.x * 256 + threadIdx.x;   // < BDIM*NN
  float s = 0.f;
#pragma unroll
  for (int p = 0; p < ZSPLIT; ++p) s += zp[(size_t)p * (BDIM * NN) + i];
  z[i] = s;
}

// ---------------- softmax over N=1024, emit g in bf16 ----------------
__global__ __launch_bounds__(256) void softmax_z_kernel(const float* __restrict__ z,
                                                        unsigned short* __restrict__ gb) {
  __shared__ float red[256];
  const int b = blockIdx.x, t = threadIdx.x;
  const float* zr = z + (size_t)b * NN;
  float mx = -1e30f;
  for (int i = t; i < NN; i += 256) mx = fmaxf(mx, zr[i]);
  red[t] = mx; __syncthreads();
  for (int s = 128; s > 0; s >>= 1) { if (t < s) red[t] = fmaxf(red[t], red[t + s]); __syncthreads(); }
  mx = red[0]; __syncthreads();
  float sum = 0.f;
  for (int i = t; i < NN; i += 256) sum += __expf(zr[i] - mx);
  red[t] = sum; __syncthreads();
  for (int s = 128; s > 0; s >>= 1) { if (t < s) red[t] += red[t + s]; __syncthreads(); }
  const float inv = 1.0f / red[0];
  for (int i = t; i < NN; i += 256) gb[(size_t)b * NN + i] = f2bf(__expf(zr[i] - mx) * inv);
}

// -------- GEMM2: y[b,ko] = sum_n g[b,n] * W[n,ko]; W tiles staged to LDS via TDM --------
// Block: 256 threads (8 waves). Block tile: ALL 64 b-rows x 128 ko-cols; wave w owns
// 16 cols; the B fragment (one LDS column gather) is reused across 4 WMMAs (4 b-tiles).
// Double-buffered tensor_load_to_lds issued by wave 0; consumers gated by
// s_wait_tensorcnt + workgroup barrier.
__global__ __launch_bounds__(256) void gemm_y_kernel(const unsigned short* __restrict__ Wb,
                                                     const unsigned short* __restrict__ Gb,
                                                     float* __restrict__ y) {
  __shared__ __align__(16) unsigned short tile[2][32 * YTN];   // 2 x 8 KB
  const int t = threadIdx.x;
  const int lane = t & 31, wave = t >> 5;
  const int m15 = lane & 15, hi = lane >> 4;
  const int ko0 = blockIdx.x * YTN;
  const unsigned lds0 = lds_off_u32(&tile[0][0]);
  const unsigned TILE_BYTES = 32 * YTN * 2;

  if (wave == 0) tdm_load_w_tile(lds0, Wb + ko0);

  v8f acc[4] = {};
  for (int i = 0; i < NN / 32; ++i) {
    const int cur = i & 1;
    if (wave == 0) __builtin_amdgcn_s_wait_tensorcnt(0);   // chunk i landed in LDS
    __syncthreads();
    if (wave == 0 && i + 1 < NN / 32)                      // prefetch next chunk (other buffer)
      tdm_load_w_tile(lds0 + (unsigned)(1 - cur) * TILE_BYTES,
                      Wb + (size_t)(i + 1) * 32 * KF + ko0);
    const int kc = i * 32;
    v16us bm;
    const unsigned short* lp = &tile[cur][(size_t)(hi * 16) * YTN + wave * 16 + m15];
#pragma unroll
    for (int e = 0; e < 16; ++e) bm[e] = lp[e * YTN];      // B column from LDS (once)
    const v16bf bf = __builtin_bit_cast(v16bf, bm);
#pragma unroll
    for (int bb = 0; bb < 4; ++bb) {                       // reuse B across 4 row-tiles of g
      const unsigned short* arow = Gb + (size_t)(bb * 16 + m15) * NN + kc;
      v16us a;
      a = *(const v16us*)(arow + hi * 16);
      acc[bb] = __builtin_amdgcn_wmma_f32_16x16x32_bf16(
          false, __builtin_bit_cast(v16bf, a), false, bf,
          (short)0, acc[bb], false, false);
    }
    __syncthreads();                                       // done reading tile[cur]
  }
  const int yk = ko0 + wave * 16;
#pragma unroll
  for (int bb = 0; bb < 4; ++bb)
#pragma unroll
    for (int r = 0; r < 8; ++r)
      y[(size_t)(bb * 16 + r + 8 * hi) * KF + yk + m15] = acc[bb][r];
}

// ---------------- conv(x_prev, w) accumulated into y ----------------
__global__ __launch_bounds__(256) void conv_add_kernel(const float* __restrict__ x,
                                                       const float* __restrict__ wc,
                                                       float* __restrict__ y) {
  __shared__ float ws[CX * KKK * KKK];            // 400 floats
  const int b = blockIdx.x / CY, cy = blockIdx.x % CY;
  const int t = threadIdx.x;
  for (int i = t; i < CX * KKK * KKK; i += 256) ws[i] = wc[(size_t)cy * CX * KKK * KKK + i];
  __syncthreads();
  const float* xb = x + (size_t)b * CX * LL * LL;
  float* yb = y + (size_t)b * KF + (size_t)cy * POSN;
  for (int pos = t; pos < POSN; pos += 256) {
    const int i = pos / MMM, j = pos % MMM;
    float s = 0.f;
    for (int cx = 0; cx < CX; ++cx) {
      const float* xp = xb + cx * LL * LL + i * LL + j;
      const float* wp = ws + cx * KKK * KKK;
#pragma unroll
      for (int ki = 0; ki < KKK; ++ki)
#pragma unroll
        for (int kj = 0; kj < KKK; ++kj)
          s += xp[ki * LL + kj] * wp[ki * KKK + kj];
    }
    yb[pos] += s;
  }
}

// ---------------- x_new = convT(f_prev, w) (adjoint of conv) ----------------
__global__ __launch_bounds__(256) void convT_kernel(const float* __restrict__ f,
                                                    const float* __restrict__ wc,
                                                    float* __restrict__ xn) {
  __shared__ float ws[CY * KKK * KKK];            // 1600 floats
  const int b = blockIdx.x / CX, cx = blockIdx.x % CX;
  const int t = threadIdx.x;
  for (int i = t; i < CY * KKK * KKK; i += 256)
    ws[i] = wc[(size_t)(i / (KKK * KKK)) * CX * KKK * KKK + (size_t)cx * KKK * KKK + (i % (KKK * KKK))];
  __syncthreads();
  const float* fb = f + (size_t)b * KF;
  float* xb = xn + (size_t)b * CX * LL * LL + (size_t)cx * LL * LL;
  for (int pos = t; pos < LL * LL; pos += 256) {
    const int p = pos / LL, q = pos % LL;
    int ki0 = p - (MMM - 1); if (ki0 < 0) ki0 = 0;
    int ki1 = p;             if (ki1 > KKK - 1) ki1 = KKK - 1;
    int kj0 = q - (MMM - 1); if (kj0 < 0) kj0 = 0;
    int kj1 = q;             if (kj1 > KKK - 1) kj1 = KKK - 1;
    float s = 0.f;
    for (int cy = 0; cy < CY; ++cy) {
      const float* fp = fb + cy * POSN;
      const float* wp = ws + cy * KKK * KKK;
      for (int ki = ki0; ki <= ki1; ++ki) {
        const int ii = p - ki;
        for (int kj = kj0; kj <= kj1; ++kj)
          s += fp[ii * MMM + (q - kj)] * wp[ki * KKK + kj];
      }
    }
    xb[pos] = s;
  }
}

// ---------------- channel softmax of y + EMA into f (in place) ----------------
__global__ __launch_bounds__(256) void f_update_kernel(const float* __restrict__ y,
                                                       float* __restrict__ f) {
  const int idx = blockIdx.x * 256 + threadIdx.x;   // (b, pos)
  if (idx >= BDIM * POSN) return;
  const int b = idx / POSN, pos = idx % POSN;
  const float* yp = y + (size_t)b * KF + pos;
  float v[CY];
  float mx = -1e30f;
#pragma unroll
  for (int c = 0; c < CY; ++c) { v[c] = yp[(size_t)c * POSN]; mx = fmaxf(mx, v[c]); }
  float sum = 0.f;
#pragma unroll
  for (int c = 0; c < CY; ++c) { v[c] = __expf(v[c] - mx); sum += v[c]; }
  const float inv = 1.0f / sum;
  float* fp = f + (size_t)b * KF + pos;
#pragma unroll
  for (int c = 0; c < CY; ++c) {
    const float fprev = fp[(size_t)c * POSN];
    fp[(size_t)c * POSN] = (1.0f - ETAY) * fprev + ETAY * (v[c] * inv);
  }
}

// ---------------- x EMA in place ----------------
__global__ __launch_bounds__(256) void x_update_kernel(float* __restrict__ x,
                                                       const float* __restrict__ xn, int n) {
  const int i = blockIdx.x * 256 + threadIdx.x;
  if (i < n) x[i] = (1.0f - ETAX) * x[i] + ETAX * xn[i];
}

// ---------------- pack outputs: [x (1048576) | f (3211264)] ----------------
__global__ __launch_bounds__(256) void pack_kernel(const float* __restrict__ x,
                                                   const float* __restrict__ f,
                                                   float* __restrict__ out) {
  const int nx = BDIM * CX * LL * LL;
  const int nf = BDIM * KF;
  const int i = blockIdx.x * 256 + threadIdx.x;
  if (i < nx) out[i] = x[i];
  else if (i < nx + nf) out[i] = f[i - nx];
}

extern "C" void kernel_launch(void* const* d_in, const int* in_sizes, int n_in,
                              void* d_out, int out_size, void* d_ws, size_t ws_size,
                              hipStream_t stream) {
  (void)in_sizes; (void)n_in; (void)out_size; (void)ws_size;
  const float* x0    = (const float*)d_in[0];
  const float* f0    = (const float*)d_in[1];
  const float* wconv = (const float*)d_in[2];
  const float* W     = (const float*)d_in[3];

  char* ws = (char*)d_ws;
  size_t off = 0;
  auto alloc = [&](size_t bytes) -> void* {
    void* p = ws + off;
    off = (off + bytes + 255) & ~(size_t)255;
    return p;
  };
  const int NX = BDIM * CX * LL * LL;   // 1,048,576
  const int NF = BDIM * KF;             // 3,211,264
  const int NW = NN * KF;               // 51,380,224

  unsigned short* Wb   = (unsigned short*)alloc((size_t)NW * 2);   // ~103 MB, L2-resident
  float*          fcur = (float*)        alloc((size_t)NF * 4);
  float*          xcur = (float*)        alloc((size_t)NX * 4);
  unsigned short* fbf  = (unsigned short*)alloc((size_t)NF * 2);
  unsigned short* gbf  = (unsigned short*)alloc((size_t)BDIM * NN * 2);
  float*          z    = (float*)        alloc((size_t)BDIM * NN * 4);
  float*          zprt = (float*)        alloc((size_t)ZSPLIT * BDIM * NN * 4);  // 4 MB
  float*          y    = (float*)        alloc((size_t)NF * 4);
  float*          xnew = (float*)        alloc((size_t)NX * 4);

  // one-time setup
  cvt_bf_kernel  <<<(NW + 255) / 256, 256, 0, stream>>>(W, Wb, NW);
  copy_f32_kernel<<<(NX + 255) / 256, 256, 0, stream>>>(x0, xcur, NX);
  copy_f32_kernel<<<(NF + 255) / 256, 256, 0, stream>>>(f0, fcur, NF);

  for (int s = 0; s < STEPS; ++s) {
    cvt_bf_kernel   <<<(NF + 255) / 256, 256, 0, stream>>>(fcur, fbf, NF);
    gemm_z_kernel   <<<dim3(BDIM / 16, NN / 16, ZSPLIT), 32, 0, stream>>>(Wb, fbf, zprt);
    reduce_z_kernel <<<(BDIM * NN) / 256, 256, 0, stream>>>(zprt, z);
    softmax_z_kernel<<<BDIM, 256, 0, stream>>>(z, gbf);
    gemm_y_kernel   <<<dim3(KF / YTN), 256, 0, stream>>>(Wb, gbf, y);
    conv_add_kernel <<<BDIM * CY, 256, 0, stream>>>(xcur, wconv, y);
    convT_kernel    <<<BDIM * CX, 256, 0, stream>>>(fcur, wconv, xnew);
    f_update_kernel <<<(BDIM * POSN) / 256, 256, 0, stream>>>(y, fcur);
    x_update_kernel <<<(NX + 255) / 256, 256, 0, stream>>>(xcur, xnew, NX);
  }

  pack_kernel<<<(NX + NF + 255) / 256, 256, 0, stream>>>(xcur, fcur, (float*)d_out);
}
